// EdgeAwareGraphLayer_27960237097201
// MI455X (gfx1250) — compile-verified, matrix operand order checked by reading, not run on hardware
//
#include <hip/hip_runtime.h>
#include <hip/hip_bf16.h>

typedef __attribute__((ext_vector_type(16))) __bf16 v16bf;
typedef __attribute__((ext_vector_type(8)))  __bf16 v8bf;
typedef __attribute__((ext_vector_type(8)))  float  v8f;

constexpr int kN = 20000;   // nodes
constexpr int kE = 200000;  // edges
constexpr int kD = 256;     // dim
constexpr int kT = 16;      // edge types

__device__ __forceinline__ float gelu_exact(float x) {
    return 0.5f * x * (1.0f + erff(x * 0.70710678118654752f));
}

// ---------------- small prep kernels ----------------

__global__ void k_f32_to_bf16(const float* __restrict__ src, __bf16* __restrict__ dst, int n) {
    int i = blockIdx.x * blockDim.x + threadIdx.x;
    if (i < n) dst[i] = (__bf16)src[i];
}

__global__ void k_zero(float* __restrict__ p, int n) {
    int i = blockIdx.x * blockDim.x + threadIdx.x;
    if (i < n) p[i] = 0.0f;
}

// Wcat (256 x 768, bf16): cols [0,256)=msg_W, [256,512)=att_W1 rows 0:256, [512,768)=att_W1 rows 256:512
__global__ void k_build_wcat(const float* __restrict__ msgW, const float* __restrict__ attW1,
                             __bf16* __restrict__ wcat) {
    int i = blockIdx.x * blockDim.x + threadIdx.x;
    if (i >= 256 * 768) return;
    int k = i / 768, n = i % 768;
    float v;
    if (n < 256)       v = msgW[k * 256 + n];
    else if (n < 512)  v = attW1[k * 256 + (n - 256)];
    else               v = attW1[(256 + k) * 256 + (n - 512)];
    wcat[i] = (__bf16)v;
}

// Per-edge-type tables: msgE[t] = e_t@msg_W + msg_b ; attT[t] = e_t@A3 + task@A4 + att_b1
__global__ void k_type_tables(const float* __restrict__ eemb, const float* __restrict__ msgW,
                              const float* __restrict__ msgb, const float* __restrict__ attW1,
                              const float* __restrict__ ab1, const float* __restrict__ task,
                              float* __restrict__ msgE, float* __restrict__ attT) {
    int t = blockIdx.x;      // 0..15
    int j = threadIdx.x;     // 0..255
    const float* e = eemb + t * 256;
    float sm = msgb[j];
    float sa = ab1[j];
    for (int k = 0; k < 256; ++k) {
        float ek = e[k];
        sm += ek * msgW[k * 256 + j];
        sa += ek * attW1[(512 + k) * 256 + j];
    }
    for (int k = 0; k < 256; ++k) sa += task[k] * attW1[(768 + k) * 256 + j];
    msgE[t * 256 + j] = sm;
    attT[t * 256 + j] = sa;
}

// Task FiLM MLPs (msg & upd): block 0 -> msg, block 1 -> upd. 512 threads.
__global__ void k_film(const float* __restrict__ task,
                       const float* __restrict__ maW1, const float* __restrict__ mab1,
                       const float* __restrict__ maW2, const float* __restrict__ mab2,
                       const float* __restrict__ uaW1, const float* __restrict__ uab1,
                       const float* __restrict__ uaW2, const float* __restrict__ uab2,
                       float* __restrict__ gm, float* __restrict__ bm,
                       float* __restrict__ gu, float* __restrict__ bu) {
    __shared__ float h[512];
    int sel = blockIdx.x;
    int j = threadIdx.x;
    const float* W1 = sel ? uaW1 : maW1;
    const float* b1 = sel ? uab1 : mab1;
    const float* W2 = sel ? uaW2 : maW2;
    const float* b2 = sel ? uab2 : mab2;
    float s = b1[j];
    for (int k = 0; k < 256; ++k) s += task[k] * W1[k * 512 + j];
    h[j] = gelu_exact(s);
    __syncthreads();
    float o = b2[j];
    for (int k = 0; k < 512; ++k) o += h[k] * W2[k * 512 + j];
    float* ge = sel ? gu : gm;
    float* be = sel ? bu : bm;
    if (j < 256) ge[j] = 1.0f + 0.5f * tanhf(o);
    else         be[j - 256] = o;
}

// ---------------- tiled bf16 WMMA GEMM: Cout = A@B (+ Cin) ----------------
// Block tile 128x128, BK=32, double-buffered LDS, 8 waves (wave tile 32x64).
// A: M x K (bf16, lda), B: K x N (bf16, ldb), Cout fp32 (ldc). N % 128 == 0, K % 32 == 0.
__global__ __launch_bounds__(256) void k_wmma_gemm(
    const __bf16* __restrict__ A, const __bf16* __restrict__ B,
    const float* __restrict__ Cin, float* __restrict__ Cout,
    int M, int N, int K, int lda, int ldb, int ldc) {
    __shared__ __bf16 sA[2][128 * 32];   // row-major [m][k]
    __shared__ __bf16 sB[2][128 * 32];   // transposed [n][k]

    const int tid  = threadIdx.x;
    const int lane = tid & 31;
    const int wave = tid >> 5;
    const int wm   = (wave & 3) * 32;   // wave row offset  (4 waves in M)
    const int wn   = (wave >> 2) * 64;  // wave col offset  (2 waves in N)
    const int rowBase = blockIdx.x * 128;
    const int colBase = blockIdx.y * 128;

    const int mrow  = lane & 15;
    const int mhalf = lane >> 4;

    // A-fill indices: 512 16B chunks (128 rows x 4 chunks), 2 per thread
    const int ar0 = tid >> 1;               // via chunk = tid + i*256 below
    // B-fill indices: thread owns 16 consecutive elems of the 32x128 tile
    const int bk  = tid >> 3;               // (tid*16)>>7 : k row 0..31
    const int bn0 = (tid & 7) * 16;         // n offset 0..112
    (void)ar0;

    v8f acc[2][4];
    v8f vzero = {0.f, 0.f, 0.f, 0.f, 0.f, 0.f, 0.f, 0.f};
    #pragma unroll
    for (int mi = 0; mi < 2; ++mi)
        #pragma unroll
        for (int ni = 0; ni < 4; ++ni) acc[mi][ni] = vzero;

    const int nk = K >> 5;

    // ---- stage tile 0 ----
    {
        #pragma unroll
        for (int i = 0; i < 2; ++i) {
            int chunk = tid + i * 256;
            int r  = chunk >> 2;
            int kc = (chunk & 3) * 8;
            int grow = rowBase + r;
            if (grow >= M) grow = M - 1;
            *(v8bf*)&sA[0][r * 32 + kc] = *(const v8bf*)&A[(size_t)grow * lda + kc];
        }
        v8bf v0 = *(const v8bf*)&B[(size_t)bk * ldb + colBase + bn0];
        v8bf v1 = *(const v8bf*)&B[(size_t)bk * ldb + colBase + bn0 + 8];
        #pragma unroll
        for (int j = 0; j < 8; ++j) {
            sB[0][(bn0 + j) * 32 + bk]     = v0[j];
            sB[0][(bn0 + 8 + j) * 32 + bk] = v1[j];
        }
    }
    __syncthreads();

    for (int kt = 0; kt < nk; ++kt) {
        const int cur = kt & 1;
        // ---- stage next tile into the other buffer (overlaps with WMMA below) ----
        if (kt + 1 < nk) {
            const int k0 = (kt + 1) * 32;
            #pragma unroll
            for (int i = 0; i < 2; ++i) {
                int chunk = tid + i * 256;
                int r  = chunk >> 2;
                int kc = (chunk & 3) * 8;
                int grow = rowBase + r;
                if (grow >= M) grow = M - 1;
                *(v8bf*)&sA[cur ^ 1][r * 32 + kc] =
                    *(const v8bf*)&A[(size_t)grow * lda + k0 + kc];
            }
            v8bf v0 = *(const v8bf*)&B[(size_t)(k0 + bk) * ldb + colBase + bn0];
            v8bf v1 = *(const v8bf*)&B[(size_t)(k0 + bk) * ldb + colBase + bn0 + 8];
            #pragma unroll
            for (int j = 0; j < 8; ++j) {
                sB[cur ^ 1][(bn0 + j) * 32 + bk]     = v0[j];
                sB[cur ^ 1][(bn0 + 8 + j) * 32 + bk] = v1[j];
            }
            if (kt + 2 < nk)
                __builtin_prefetch(&A[(size_t)(rowBase + (tid >> 1)) * lda + k0 + 32], 0, 1);
        }

        // ---- compute on current buffer ----
        // A/B fragments per ISA 16-bit 16x32 layout:
        // lane half 0: K {0..7,16..23}; half 1: K {8..15,24..31}
        v16bf af[2], bfr[4];
        #pragma unroll
        for (int mi = 0; mi < 2; ++mi) {
            const __bf16* ap = &sA[cur][(wm + mi * 16 + mrow) * 32 + mhalf * 8];
            v8bf c0 = *(const v8bf*)ap;
            v8bf c1 = *(const v8bf*)(ap + 16);
            v16bf a;
            #pragma unroll
            for (int j = 0; j < 8; ++j) { a[j] = c0[j]; a[8 + j] = c1[j]; }
            af[mi] = a;
        }
        #pragma unroll
        for (int ni = 0; ni < 4; ++ni) {
            const __bf16* bp = &sB[cur][(wn + ni * 16 + mrow) * 32 + mhalf * 8];
            v8bf c0 = *(const v8bf*)bp;
            v8bf c1 = *(const v8bf*)(bp + 16);
            v16bf b;
            #pragma unroll
            for (int j = 0; j < 8; ++j) { b[j] = c0[j]; b[8 + j] = c1[j]; }
            bfr[ni] = b;
        }
        #pragma unroll
        for (int mi = 0; mi < 2; ++mi)
            #pragma unroll
            for (int ni = 0; ni < 4; ++ni)
                acc[mi][ni] = __builtin_amdgcn_wmma_f32_16x16x32_bf16(
                    false, af[mi], false, bfr[ni], (short)0, acc[mi][ni], false, false);

        __syncthreads();   // next iteration reads buf^1 / refills buf
    }

    // epilogue: C/D layout — VGPR j: lanes0-15 M=j, lanes16-31 M=j+8; N=lane&15
    #pragma unroll
    for (int mi = 0; mi < 2; ++mi)
        #pragma unroll
        for (int ni = 0; ni < 4; ++ni) {
            int col = colBase + wn + ni * 16 + (lane & 15);
            #pragma unroll
            for (int j = 0; j < 8; ++j) {
                int row = rowBase + wm + mi * 16 + j + 8 * (lane >> 4);
                if (row < M) {
                    size_t idx = (size_t)row * ldc + col;
                    float v = acc[mi][ni][j];
                    if (Cin) v += Cin[idx];
                    Cout[idx] = v;
                }
            }
        }
}

// ---------------- per-edge kernel: one wave per edge ----------------
// Y: N x 768 = [P|Q|R]. Computes attention weight, scatters weighted message + degree.
__global__ __launch_bounds__(256) void k_edge(
    const int* __restrict__ src, const int* __restrict__ dst, const int* __restrict__ etype,
    const float* __restrict__ Y, const float* __restrict__ msgE, const float* __restrict__ attT,
    const float* __restrict__ attW2, const float* __restrict__ ab2,
    const float* __restrict__ esc,
    float* __restrict__ S, float* __restrict__ deg, int E) {
    int w = blockIdx.x * 8 + (threadIdx.x >> 5);
    if (w >= E) return;
    int lane = threadIdx.x & 31;
    int s = src[w], d = dst[w], t = etype[w];
    int c = lane * 2;  // float4 index (each lane owns 8 contiguous floats)

    float q[8], r[8], at[8], w2[8];
    const float4* Qp = (const float4*)(Y + (size_t)s * 768 + 256);
    const float4* Rp = (const float4*)(Y + (size_t)d * 768 + 512);
    const float4* Ap = (const float4*)(attT + t * 256);
    const float4* Wp = (const float4*)attW2;
    *(float4*)&q[0]  = Qp[c]; *(float4*)&q[4]  = Qp[c + 1];
    *(float4*)&r[0]  = Rp[c]; *(float4*)&r[4]  = Rp[c + 1];
    *(float4*)&at[0] = Ap[c]; *(float4*)&at[4] = Ap[c + 1];
    *(float4*)&w2[0] = Wp[c]; *(float4*)&w2[4] = Wp[c + 1];

    float part = 0.f;
    #pragma unroll
    for (int j = 0; j < 8; ++j) part += gelu_exact(q[j] + r[j] + at[j]) * w2[j];
    #pragma unroll
    for (int off = 16; off > 0; off >>= 1) part += __shfl_xor(part, off, 32);
    float wgt = 1.0f + esc[0] * tanhf(part + ab2[0]);

    float pv[8], me[8];
    const float4* Pp = (const float4*)(Y + (size_t)s * 768);
    const float4* Mp = (const float4*)(msgE + t * 256);
    *(float4*)&pv[0] = Pp[c]; *(float4*)&pv[4] = Pp[c + 1];
    *(float4*)&me[0] = Mp[c]; *(float4*)&me[4] = Mp[c + 1];

    float* Sd = S + (size_t)d * 256 + lane * 8;
    #pragma unroll
    for (int j = 0; j < 8; ++j) atomicAdd(&Sd[j], (pv[j] + me[j]) * wgt);
    if (lane == 0) atomicAdd(&deg[d], wgt);
}

// agg normalize + both FiLMs folded; emit bf16 for the update GEMM
__global__ void k_node_prep(const float* __restrict__ S, const float* __restrict__ deg,
                            const float* __restrict__ gm, const float* __restrict__ bm,
                            const float* __restrict__ gu, const float* __restrict__ bu,
                            __bf16* __restrict__ Ab, int n) {
    int i = blockIdx.x * blockDim.x + threadIdx.x;
    if (i >= n) return;
    int j = i & 255, row = i >> 8;
    float dg = deg[row];
    float a = (gm[j] * S[i] + bm[j] * dg) / fmaxf(dg, 1.0f);
    Ab[i] = (__bf16)(a * gu[j] + bu[j]);
}

__global__ void k_gelu_bias_bf16(const float* __restrict__ T, const float* __restrict__ b,
                                 __bf16* __restrict__ H, int n) {
    int i = blockIdx.x * blockDim.x + threadIdx.x;
    if (i >= n) return;
    H[i] = (__bf16)gelu_exact(T[i] + b[i & 255]);
}

// y = x + T2 + up_b2 ; LayerNorm(y)*g + b. One row per block (256 threads).
__global__ __launch_bounds__(256) void k_final_ln(
    const float* __restrict__ x, const float* __restrict__ T2, const float* __restrict__ b2,
    const float* __restrict__ g, const float* __restrict__ b, float* __restrict__ out) {
    __shared__ float red[256];
    int row = blockIdx.x, t = threadIdx.x;
    size_t idx = (size_t)row * 256 + t;
    float v = x[idx] + T2[idx] + b2[t];
    red[t] = v; __syncthreads();
    for (int s = 128; s > 0; s >>= 1) { if (t < s) red[t] += red[t + s]; __syncthreads(); }
    float mu = red[0] * (1.0f / 256.0f); __syncthreads();
    float dv = v - mu;
    red[t] = dv * dv; __syncthreads();
    for (int s = 128; s > 0; s >>= 1) { if (t < s) red[t] += red[t + s]; __syncthreads(); }
    float var = red[0] * (1.0f / 256.0f);
    out[idx] = dv * rsqrtf(var + 1e-5f) * g[t] + b[t];
}

// ---------------- host ----------------

extern "C" void kernel_launch(void* const* d_in, const int* in_sizes, int n_in,
                              void* d_out, int out_size, void* d_ws, size_t ws_size,
                              hipStream_t stream) {
    const float* node = (const float*)d_in[0];
    const int*   eidx = (const int*)d_in[1];
    const int*   etyp = (const int*)d_in[2];
    const float* task = (const float*)d_in[3];
    const float* eemb = (const float*)d_in[4];
    const float* msgW = (const float*)d_in[5];
    const float* msgb = (const float*)d_in[6];
    const float* aW1  = (const float*)d_in[7];
    const float* ab1  = (const float*)d_in[8];
    const float* aW2  = (const float*)d_in[9];
    const float* ab2  = (const float*)d_in[10];
    const float* esc  = (const float*)d_in[11];
    const float* maW1 = (const float*)d_in[12];
    const float* mab1 = (const float*)d_in[13];
    const float* maW2 = (const float*)d_in[14];
    const float* mab2 = (const float*)d_in[15];
    const float* upW1 = (const float*)d_in[16];
    const float* upb1 = (const float*)d_in[17];
    const float* upW2 = (const float*)d_in[18];
    const float* upb2 = (const float*)d_in[19];
    const float* uaW1 = (const float*)d_in[20];
    const float* uab1 = (const float*)d_in[21];
    const float* uaW2 = (const float*)d_in[22];
    const float* uab2 = (const float*)d_in[23];
    const float* lng  = (const float*)d_in[24];
    const float* lnb  = (const float*)d_in[25];
    float* out = (float*)d_out;

    const int* src = eidx;
    const int* dst = eidx + kE;

    char* w = (char*)d_ws;
    auto carve = [&](size_t bytes) -> char* {
        char* p = w; w += (bytes + 255) & ~(size_t)255; return p;
    };
    __bf16* Xb    = (__bf16*)carve((size_t)kN * kD * 2);      // bf16 node emb
    __bf16* Wcat  = (__bf16*)carve((size_t)kD * 768 * 2);     // [msg_W|A1|A2]
    __bf16* U1b   = (__bf16*)carve((size_t)512 * 256 * 2);    // up_W1 bf16
    __bf16* U2b   = (__bf16*)carve((size_t)256 * 256 * 2);    // up_W2 bf16
    float*  msgE  = (float*)carve((size_t)kT * kD * 4);
    float*  attT  = (float*)carve((size_t)kT * kD * 4);
    float*  gm    = (float*)carve(256 * 4);
    float*  bm    = (float*)carve(256 * 4);
    float*  gu    = (float*)carve(256 * 4);
    float*  bu    = (float*)carve(256 * 4);
    float*  degv  = (float*)carve((size_t)kN * 4);
    float*  Sbuf  = (float*)carve((size_t)kN * kD * 4);
    __bf16* Ab    = (__bf16*)carve((size_t)kN * kD * 2);
    float*  Y     = (float*)carve((size_t)kN * 768 * 4);      // P|Q|R, reused below
    // reuse Y region after edge phase:
    float*  T   = Y;                                          // 20.48 MB
    float*  T2  = Y + (size_t)kN * kD;                        // 20.48 MB
    __bf16* Hb  = (__bf16*)(Y + (size_t)2 * kN * kD);         // 10.24 MB

    const int nElem = kN * kD;  // 5,120,000

    // prep
    k_f32_to_bf16<<<(nElem + 255) / 256, 256, 0, stream>>>(node, Xb, nElem);
    k_build_wcat<<<(256 * 768 + 255) / 256, 256, 0, stream>>>(msgW, aW1, Wcat);
    k_f32_to_bf16<<<(512 * 256 + 255) / 256, 256, 0, stream>>>(upW1, U1b, 512 * 256);
    k_f32_to_bf16<<<(256 * 256 + 255) / 256, 256, 0, stream>>>(upW2, U2b, 256 * 256);
    k_type_tables<<<kT, 256, 0, stream>>>(eemb, msgW, msgb, aW1, ab1, task, msgE, attT);
    k_film<<<2, 512, 0, stream>>>(task, maW1, mab1, maW2, mab2, uaW1, uab1, uaW2, uab2,
                                  gm, bm, gu, bu);
    k_zero<<<(nElem + 255) / 256, 256, 0, stream>>>(Sbuf, nElem);
    k_zero<<<(kN + 255) / 256, 256, 0, stream>>>(degv, kN);

    // per-node projections: Y = Xb @ Wcat  (20000x256 @ 256x768)
    dim3 g1((kN + 127) / 128, 768 / 128);
    k_wmma_gemm<<<g1, 256, 0, stream>>>(Xb, Wcat, nullptr, Y, kN, 768, 256, 256, 768, 768);

    // edge phase: attention + weighted scatter
    k_edge<<<(kE + 7) / 8, 256, 0, stream>>>(src, dst, etyp, Y, msgE, attT, aW2, ab2, esc,
                                             Sbuf, degv, kE);

    // node aggregation epilogue -> bf16
    k_node_prep<<<(nElem + 255) / 256, 256, 0, stream>>>(Sbuf, degv, gm, bm, gu, bu, Ab, nElem);

    // update MLP: T = Xb@up_W1[0:256]  then  T += Ab@up_W1[256:512]
    dim3 g2((kN + 127) / 128, 256 / 128);
    k_wmma_gemm<<<g2, 256, 0, stream>>>(Xb, U1b, nullptr, T, kN, 256, 256, 256, 256, 256);
    k_wmma_gemm<<<g2, 256, 0, stream>>>(Ab, U1b + 256 * 256, T, T, kN, 256, 256, 256, 256, 256);
    k_gelu_bias_bf16<<<(nElem + 255) / 256, 256, 0, stream>>>(T, upb1, Hb, nElem);
    k_wmma_gemm<<<g2, 256, 0, stream>>>(Hb, U2b, nullptr, T2, kN, 256, 256, 256, 256, 256);

    // residual + LayerNorm
    k_final_ln<<<kN, 256, 0, stream>>>(node, T2, upb2, lng, lnb, out);
}
